// SymmetricCurve1DConvV2_26834955666001
// MI455X (gfx1250) — compile-verified, matrix operand order checked by reading, not run on hardware
//
#include <hip/hip_runtime.h>
#include <hip/hip_bf16.h>

typedef __attribute__((ext_vector_type(16))) _Float16 v16h;
typedef __attribute__((ext_vector_type(8)))  _Float16 v8h;
typedef __attribute__((ext_vector_type(8)))  float    v8f;
typedef __attribute__((ext_vector_type(4))) unsigned int u32x4;
typedef __attribute__((ext_vector_type(8))) int          i32x8;
typedef __attribute__((ext_vector_type(4))) int          i32x4;

#define NPTS   262144
#define CLEN   256
#define NCURVE 1024
#define PADW   6
#define NCONV  268294          /* NPTS + (NCURVE+1)*PADW */
#define CSTR   262             /* PADW + CLEN, per-curve stride in padded axis */
#define TOFF   16              /* front halo per row (zeroed) so TDM never reads t<0 */
#define LDT    268352          /* row length: TOFF + NCONV + tail halo, mult of 32 */
#define CIN    256
#define KTOT   768             /* 3 taps * 256 channels */
#define LT     32              /* positions per conv tile */
#define CHALF  128             /* channels staged per phase */
#define SSTR   136             /* CHALF + 8, keeps 16B alignment (8*17) */
#define TDMW   40              /* staged window width: t0-2 .. t0+37 (80B rows) */

static __device__ __forceinline__ float lrelu(float v) { return v >= 0.f ? v : 0.01f * v; }

// ---------------------------------------------------------------------------
// TDM: DMA a 2-D tile (CHALF rows x TDMW f16) of a [C][LDT] tensor into LDS.
// D# per CDNA5 ISA 8.3/8.4: group0 = {count, lds_addr, global_addr, type=2},
// group1 = {data_size=2B, tensor_dim0=LDT, tensor_dim1=CHALF, tile 40x128,
//           tensor_dim0_stride=LDT}.  Groups 2/3 unused (2-D tile).
// ---------------------------------------------------------------------------
static __device__ __forceinline__ void tdm_load_tile(const _Float16* gsrc, void* ldsDst) {
  unsigned long long ga = (unsigned long long)(uintptr_t)gsrc;
  unsigned lds = (unsigned)(uintptr_t)ldsDst;   // generic addr low bits = LDS offset
  u32x4 g0;
  g0[0] = 1u;                                                  // count=1, user D#
  g0[1] = lds;                                                 // lds_addr
  g0[2] = (unsigned)ga;                                        // global_addr[31:0]
  g0[3] = (unsigned)((ga >> 32) & 0x01FFFFFFull) | 0x80000000u; // ga[56:32] | type=2
  i32x8 g1;
  g1[0] = 0x00010000;                          // workgroup_mask=0, data_size=1 (2B)
  g1[1] = (int)((LDT & 0xFFFF) << 16);         // tensor_dim0[15:0]
  g1[2] = (int)((LDT >> 16) | (CHALF << 16));  // tensor_dim0[31:16] | tensor_dim1 lo
  g1[3] = (int)(TDMW << 16);                   // tensor_dim1 hi (0) | tile_dim0
  g1[4] = (int)CHALF;                          // tile_dim1 | tile_dim2=0
  g1[5] = (int)LDT;                            // tensor_dim0_stride lo32
  g1[6] = 0;                                   // stride hi | tensor_dim1_stride lo
  g1[7] = 0;
  i32x4 z4;
  z4[0] = z4[1] = z4[2] = z4[3] = 0;
  i32x8 z8;
  z8[0] = z8[1] = z8[2] = z8[3] = z8[4] = z8[5] = z8[6] = z8[7] = 0;
  __builtin_amdgcn_tensor_load_to_lds(g0, g1, z4, z4, z8, 0);
}

// ---------------------------------------------------------------------------
// Repack conv_w (O, 256, 3) f32  ->  f16 [o][tap*256 + c]
// ---------------------------------------------------------------------------
__global__ __launch_bounds__(256) void prep_weights(const float* __restrict__ w,
                                                    _Float16* __restrict__ wp, int O) {
  int idx = blockIdx.x * 256 + threadIdx.x;
  if (idx >= O * KTOT) return;
  int o = idx / KTOT, kk = idx - o * KTOT;
  int tap = kk / CIN, c = kk - tap * CIN;
  wp[idx] = (_Float16)w[((size_t)o * CIN + c) * 3 + tap];
}

// ---------------------------------------------------------------------------
// Build h0 = (x || |avg fwd-diff|) scattered into padded, channel-major f16
// buffer [256][LDT] at row offset TOFF.  Pads/halos pre-zeroed by memset.
// ---------------------------------------------------------------------------
__global__ __launch_bounds__(256) void build_h0(const float* __restrict__ x,
                                                _Float16* __restrict__ h0) {
  __shared__ float xs[34][128];
  const int tid = threadIdx.x;
  const int j0 = blockIdx.x * 32;
  for (int i = tid; i < 34 * 128; i += 256) {
    int rr = i / 128, c = i - rr * 128;
    int j = j0 - 1 + rr;
    xs[rr][c] = (j >= 0 && j < NPTS) ? x[(size_t)j * 128 + c] : 0.f;
  }
  __syncthreads();
  const int tl = tid & 31, og = tid >> 5;
  const int curve = j0 / CLEN;
  const int r = (j0 % CLEN) + tl;                 // 32 | 256, stays in-curve
  const int t = PADW + curve * CSTR + r;          // dest position in padded axis
  const bool hasL = (r > 0), hasR = (r < CLEN - 1);
  const float denom = fmaxf(1.f, (float)hasL + (float)hasR);
  for (int cb = 0; cb < 32; ++cb) {
    int c = cb * 8 + og;
    float val;
    if (c < 128) {
      val = xs[tl + 1][c];
    } else {
      int cd = c - 128;
      float el = hasL ? xs[tl + 1][cd] - xs[tl][cd] : 0.f;
      float er = hasR ? xs[tl + 2][cd] - xs[tl + 1][cd] : 0.f;
      val = fabsf((el + er) / denom);
    }
    h0[(size_t)c * LDT + TOFF + t] = (_Float16)val;
  }
}

// ---------------------------------------------------------------------------
// Symmetric conv as WMMA GEMM: y[o,t] = bias[o] + sum_{tap,c} W[o,tap,c]*S_tap[c,t]
// Activation windows staged by the Tensor Data Mover (both phases issued
// up-front by wave 0, gated with s_wait_tensorcnt).  Input activation for
// layers>0, lrelu(scale*in+shift), is applied while building the symmetric-tap
// matrices.  Per-channel sum/sumsq accumulated for BatchNorm.
// Grid: (ceil(NCONV/32), O/256), 256 threads.
// ---------------------------------------------------------------------------
__global__ __launch_bounds__(256) void conv_sym_wmma(
    const _Float16* __restrict__ Wp, const float* __restrict__ bias,
    const _Float16* __restrict__ bufIn,
    const float* __restrict__ scIn, const float* __restrict__ shIn, int actIn,
    _Float16* __restrict__ bufOut, float* __restrict__ stats, int O) {
  __shared__ __align__(16) _Float16 actS[2][CHALF][TDMW];  // raw windows (TDM dest)
  __shared__ __align__(16) _Float16 S2[3][LT][SSTR];       // [tap][t][c] sym taps

  const int tid = threadIdx.x;
  const int t0 = blockIdx.x * LT;
  const int wave = tid >> 5, lane = tid & 31, lm = lane & 15, hi = lane >> 4;
  const int obase = blockIdx.y * 256 + wave * 32;

  v8f acc[2][2];
#pragma unroll
  for (int mi = 0; mi < 2; ++mi)
#pragma unroll
    for (int ni = 0; ni < 2; ++ni)
#pragma unroll
      for (int q = 0; q < 8; ++q) acc[mi][ni][q] = 0.f;

  // Kick off both phase DMAs (in-order per wave); window = [t0-2, t0+38).
  if (wave == 0) {
    const _Float16* src = bufIn + (size_t)(TOFF + t0 - 2);
    tdm_load_tile(src, &actS[0][0][0]);
    tdm_load_tile(src + (size_t)CHALF * LDT, &actS[1][0][0]);
  }

  for (int ph = 0; ph < 2; ++ph) {
    const int cbase = ph * CHALF;
    if (wave == 0) {
      if (ph == 0) __builtin_amdgcn_s_wait_tensorcnt(1);
      else         __builtin_amdgcn_s_wait_tensorcnt(0);
    }
    __syncthreads();  // publish TDM tile; also fences prior phase's S2 reads
    // Symmetric tap matrices, c-contiguous for B-fragment loads; activation
    // (per-channel affine + lrelu) applied per element read.
    for (int i = tid; i < 3 * LT * CHALF; i += 256) {
      int tap = i / (LT * CHALF);
      int r = i - tap * LT * CHALF;
      int tl = r / CHALF, cc = r - tl * CHALF;
      int w = tl + 2;
      float sc = actIn ? scIn[cbase + cc] : 1.f;
      float sh = actIn ? shIn[cbase + cc] : 0.f;
      float v;
      if (tap == 0) {
        float a = (float)actS[ph][cc][w];
        v = actIn ? lrelu(a * sc + sh) : a;
      } else {
        float a0 = (float)actS[ph][cc][w - tap];
        float a1 = (float)actS[ph][cc][w + tap];
        if (actIn) {
          a0 = lrelu(a0 * sc + sh);
          a1 = lrelu(a1 * sc + sh);
        }
        v = a0 + a1;
      }
      S2[tap][tl][cc] = (_Float16)v;
    }
    __syncthreads();

#pragma unroll 4
    for (int kc = 0; kc < 12; ++kc) {
      const int tap = kc >> 2, c0 = (kc & 3) * 32;
      union HV { v16h v; v8h h[2]; };
      HV aF[2], bF[2];
      // A fragments: 16 o-rows x 32 K, 16-bit layout (lanes<16: K 0..7,16..23;
      // lanes>=16: +8) -> two contiguous 16B loads per row.
#pragma unroll
      for (int mi = 0; mi < 2; ++mi) {
        const _Float16* wp = Wp + (size_t)(obase + mi * 16 + lm) * KTOT +
                             tap * CIN + cbase + c0 + hi * 8;
        aF[mi].h[0] = *(const v8h*)wp;
        aF[mi].h[1] = *(const v8h*)(wp + 16);
      }
      if (kc + 1 < 12) {
        __builtin_prefetch(Wp + (size_t)(obase + lm) * KTOT +
                               ((kc + 1) >> 2) * CIN + cbase + ((kc + 1) & 3) * 32,
                           0, 1);
      }
      // B fragments: 32 K x 16 t, same K striping, c contiguous in LDS.
#pragma unroll
      for (int ni = 0; ni < 2; ++ni) {
        const _Float16* sp = &S2[tap][ni * 16 + lm][c0 + hi * 8];
        bF[ni].h[0] = *(const v8h*)sp;
        bF[ni].h[1] = *(const v8h*)(sp + 16);
      }
#pragma unroll
      for (int mi = 0; mi < 2; ++mi)
#pragma unroll
        for (int ni = 0; ni < 2; ++ni)
          acc[mi][ni] = __builtin_amdgcn_wmma_f32_16x16x32_f16(
              false, aF[mi].v, false, bF[ni].v, (short)0, acc[mi][ni],
              false, false);
    }
  }

  // Epilogue: bias, store f16, and per-channel BN partial sums.
#pragma unroll
  for (int mi = 0; mi < 2; ++mi) {
#pragma unroll
    for (int v = 0; v < 8; ++v) {
      const int o = obase + mi * 16 + v + hi * 8;   // C/D row = vgpr + 8*hi
      const float bo = bias[o];
      float ys = 0.f, yq = 0.f;
#pragma unroll
      for (int ni = 0; ni < 2; ++ni) {
        const int t = t0 + ni * 16 + lm;            // C/D col = lane%16
        float y = acc[mi][ni][v] + bo;
        if (t < NCONV) {
          bufOut[(size_t)o * LDT + TOFF + t] = (_Float16)y;
          ys += y;
          yq += y * y;
        }
      }
#pragma unroll
      for (int m = 1; m < 16; m <<= 1) {
        ys += __shfl_xor(ys, m, 32);
        yq += __shfl_xor(yq, m, 32);
      }
      if (lm == 0) {
        atomicAdd(&stats[o], ys);
        atomicAdd(&stats[O + o], yq);
      }
    }
  }
}

// ---------------------------------------------------------------------------
// BN finalize: fold mean/var/gamma/beta into scale/shift (biased var, eps=1e-5).
// ---------------------------------------------------------------------------
__global__ __launch_bounds__(256) void bn_finalize(const float* __restrict__ stats,
                                                   const float* __restrict__ g,
                                                   const float* __restrict__ be,
                                                   float* __restrict__ sc,
                                                   float* __restrict__ sh, int O) {
  int o = blockIdx.x * 256 + threadIdx.x;
  if (o >= O) return;
  const float inv_n = 1.f / (float)NCONV;
  float mean = stats[o] * inv_n;
  float var = stats[O + o] * inv_n - mean * mean;
  float is = rsqrtf(var + 1e-5f);
  float s = g[o] * is;
  sc[o] = s;
  sh[o] = be[o] - mean * s;
}

// ---------------------------------------------------------------------------
// Gather: out[j, o] = lrelu(bn2(y2[o, dest(j)])), 32 points per block.
// ---------------------------------------------------------------------------
__global__ __launch_bounds__(256) void gather_out(const _Float16* __restrict__ y2,
                                                  const float* __restrict__ sc,
                                                  const float* __restrict__ sh,
                                                  float* __restrict__ out) {
  const int tid = threadIdx.x;
  const int j0 = blockIdx.x * 32;
  const int tl = tid & 31, og = tid >> 5;
  const int curve = j0 / CLEN;
  const int t = PADW + curve * CSTR + (j0 % CLEN) + tl;
  const int j = j0 + tl;
  for (int ob = 0; ob < 64; ++ob) {
    int o = ob * 8 + og;
    float v = (float)y2[(size_t)o * LDT + TOFF + t];
    v = lrelu(v * sc[o] + sh[o]);
    out[(size_t)j * 512 + o] = v;
  }
}

// ---------------------------------------------------------------------------
extern "C" void kernel_launch(void* const* d_in, const int* in_sizes, int n_in,
                              void* d_out, int out_size, void* d_ws, size_t ws_size,
                              hipStream_t stream) {
  (void)in_sizes; (void)n_in; (void)out_size; (void)ws_size;
  const float* x   = (const float*)d_in[0];
  const float* w0  = (const float*)d_in[4];
  const float* b0  = (const float*)d_in[5];
  const float* bg0 = (const float*)d_in[6];
  const float* bb0 = (const float*)d_in[7];
  const float* w1  = (const float*)d_in[8];
  const float* b1  = (const float*)d_in[9];
  const float* bg1 = (const float*)d_in[10];
  const float* bb1 = (const float*)d_in[11];
  const float* w2  = (const float*)d_in[12];
  const float* b2  = (const float*)d_in[13];
  const float* bg2 = (const float*)d_in[14];
  const float* bb2 = (const float*)d_in[15];
  float* out = (float*)d_out;

  char* ws = (char*)d_ws;
  size_t off = 0;
  auto alloc = [&](size_t bytes) -> void* {
    void* p = ws + off;
    off = (off + bytes + 255) & ~(size_t)255;
    return p;
  };
  _Float16* wp0 = (_Float16*)alloc((size_t)256 * KTOT * 2);
  _Float16* wp1 = (_Float16*)alloc((size_t)256 * KTOT * 2);
  _Float16* wp2 = (_Float16*)alloc((size_t)512 * KTOT * 2);
  float* st0 = (float*)alloc(2 * 256 * 4);
  float* st1 = (float*)alloc(2 * 256 * 4);
  float* st2 = (float*)alloc(2 * 512 * 4);
  float* sc0 = (float*)alloc(256 * 4);
  float* sh0 = (float*)alloc(256 * 4);
  float* sc1 = (float*)alloc(256 * 4);
  float* sh1 = (float*)alloc(256 * 4);
  float* sc2 = (float*)alloc(512 * 4);
  float* sh2 = (float*)alloc(512 * 4);
  _Float16* bufA = (_Float16*)alloc((size_t)CIN * LDT * 2);  // h0, reused as y1
  _Float16* bufB = (_Float16*)alloc((size_t)CIN * LDT * 2);  // y0
  _Float16* bufC = (_Float16*)alloc((size_t)512 * LDT * 2);  // y2

  // Zero pads + halos so the TDM staging path needs no per-element guards.
  (void)hipMemsetAsync(bufA, 0, (size_t)CIN * LDT * 2, stream);
  (void)hipMemsetAsync(bufB, 0, (size_t)CIN * LDT * 2, stream);
  (void)hipMemsetAsync(st0, 0, 2 * 256 * 4, stream);
  (void)hipMemsetAsync(st1, 0, 2 * 256 * 4, stream);
  (void)hipMemsetAsync(st2, 0, 2 * 512 * 4, stream);

  prep_weights<<<(256 * KTOT + 255) / 256, 256, 0, stream>>>(w0, wp0, 256);
  prep_weights<<<(256 * KTOT + 255) / 256, 256, 0, stream>>>(w1, wp1, 256);
  prep_weights<<<(512 * KTOT + 255) / 256, 256, 0, stream>>>(w2, wp2, 512);

  build_h0<<<NPTS / 32, 256, 0, stream>>>(x, bufA);

  const int NT = (NCONV + LT - 1) / LT;
  dim3 grid1(NT, 1), grid2(NT, 2);
  conv_sym_wmma<<<grid1, 256, 0, stream>>>(wp0, b0, bufA, nullptr, nullptr, 0,
                                           bufB, st0, 256);
  bn_finalize<<<1, 256, 0, stream>>>(st0, bg0, bb0, sc0, sh0, 256);
  conv_sym_wmma<<<grid1, 256, 0, stream>>>(wp1, b1, bufB, sc0, sh0, 1,
                                           bufA, st1, 256);
  bn_finalize<<<1, 256, 0, stream>>>(st1, bg1, bb1, sc1, sh1, 256);
  conv_sym_wmma<<<grid2, 256, 0, stream>>>(wp2, b2, bufA, sc1, sh1, 1,
                                           bufC, st2, 512);
  bn_finalize<<<2, 256, 0, stream>>>(st2, bg2, bb2, sc2, sh2, 512);

  gather_out<<<NPTS / 32, 256, 0, stream>>>(bufC, sc2, sh2, out);
}
